// SwinTransformerBlock_79001628443398
// MI455X (gfx1250) — matrix operand order repaired and checked by reading, not compiled, not run on hardware
//
#include <hip/hip_runtime.h>

// ---------------------------------------------------------------- types
typedef __attribute__((ext_vector_type(16))) _Float16 v16h;
typedef __attribute__((ext_vector_type(8)))  _Float16 v8h;
typedef __attribute__((ext_vector_type(8)))  float    v8f;

union FragU { v16h v; v8h h[2]; };

// A fragment (16x32 f16, M x K), contiguous K order (matches B):
// lane l: row = l&15 ; halves e -> K = 16*(l>>4)+e
__device__ __forceinline__ v16h load_a_frag(const _Float16* A, int lda, int row0, int k0, int lane) {
    const int r  = lane & 15;
    const int hh = lane >> 4;
    const _Float16* p = A + (size_t)(row0 + r) * lda + k0;
    FragU u;
    u.h[0] = *(const v8h*)(p + 16 * hh);
    u.h[1] = *(const v8h*)(p + 16 * hh + 8);
    return u.v;
}

// B fragment (32x16 f16, K x N) from a [N,K] row-major (transposed-weight) buffer:
// lane l: col = l&15 ; halves e -> K = k0 + 16*(l>>4) + e  (one contiguous 32B load)
__device__ __forceinline__ v16h load_b_frag(const _Float16* Wt, int ldb, int n0, int k0, int lane) {
    const int n  = lane & 15;
    const int hh = lane >> 4;
    return *(const v16h*)(Wt + (size_t)(n0 + n) * ldb + k0 + 16 * hh);
}

__device__ __forceinline__ v8f wmma16x16x32(v16h a, v16h b, v8f c) {
    return __builtin_amdgcn_wmma_f32_16x16x32_f16(false, a, false, b, (short)0, c, false, false);
}

// ---------------------------------------------------------------- ds_swizzle xor-butterfly reductions
// group-of-32 mode: offset = {xor_mask[14:10], or_mask[9:5]=0, and_mask[4:0]=0x1f}
#define SWZ_XOR(x, pat) __int_as_float(__builtin_amdgcn_ds_swizzle(__float_as_int(x), (pat)))

__device__ __forceinline__ float red_sum32(float s) {   // full-wave sum
    s += SWZ_XOR(s, 0x041F);  // xor 1
    s += SWZ_XOR(s, 0x081F);  // xor 2
    s += SWZ_XOR(s, 0x101F);  // xor 4
    s += SWZ_XOR(s, 0x201F);  // xor 8
    s += SWZ_XOR(s, 0x401F);  // xor 16
    return s;
}
__device__ __forceinline__ float red_max16(float x) {   // within 16-lane halves
    x = fmaxf(x, SWZ_XOR(x, 0x041F));
    x = fmaxf(x, SWZ_XOR(x, 0x081F));
    x = fmaxf(x, SWZ_XOR(x, 0x101F));
    x = fmaxf(x, SWZ_XOR(x, 0x201F));
    return x;
}
__device__ __forceinline__ float red_sum16(float x) {   // within 16-lane halves
    x += SWZ_XOR(x, 0x041F);
    x += SWZ_XOR(x, 0x081F);
    x += SWZ_XOR(x, 0x101F);
    x += SWZ_XOR(x, 0x201F);
    return x;
}

// ---------------------------------------------------------------- constants
constexpr int CB    = 32;      // batch
constexpr int CH    = 56;      // H
constexpr int CW    = 56;      // W
constexpr int CC    = 192;     // channels
constexpr int CNH   = 6;       // heads
constexpr int CHD   = 32;      // head dim
constexpr int CN    = 49;      // tokens per window
constexpr int CQKV  = 576;     // 3*C
constexpr int CHID  = 768;
constexpr int CROWS = CB * CH * CW;   // 100352 tokens = GEMM rows
constexpr int CNWIN = 64;             // windows per image (8x8)

// ---------------------------------------------------------------- weight f16 transpose: Wt[n*K+k] = W[k*N+n]
__global__ void swin_transpose_f16(const float* __restrict__ W, _Float16* __restrict__ Wt, int K, int N) {
    int i = blockIdx.x * 256 + threadIdx.x;
    if (i >= K * N) return;
    int n = i / K, k = i - n * K;
    Wt[i] = (_Float16)W[(size_t)k * N + n];
}

// ---------------------------------------------------------------- fused (rel-pos bias + shift mask), padded 49->64 with -1e30
// fused[((wm*6 + head)*64 + m)*64 + n]
__global__ void swin_fuse_bias_mask(const float* __restrict__ rel_table,
                                    const int*   __restrict__ rel_index,
                                    const float* __restrict__ maskm,
                                    float* __restrict__ fused) {
    const int i = blockIdx.x * 256 + threadIdx.x;   // < 64*6*64*64
    const int n  = i & 63;
    const int m  = (i >> 6) & 63;
    const int hw = i >> 12;
    const int head = hw % CNH;
    const int wm   = hw / CNH;
    float v = -1e30f;
    if (m < CN && n < CN)
        v = rel_table[rel_index[m * CN + n] * CNH + head] + maskm[(size_t)wm * (CN * CN) + m * CN + n];
    fused[i] = v;
}

// ---------------------------------------------------------------- LayerNorm (wave per token), optional shift+window-partition remap
template<bool REMAP>
__global__ __launch_bounds__(256) void swin_ln_f16(const float* __restrict__ in,
                                                   const float* __restrict__ g,
                                                   const float* __restrict__ b,
                                                   _Float16* __restrict__ out) {
    const int token = blockIdx.x * 8 + (threadIdx.x >> 5);
    const int lane  = threadIdx.x & 31;
    const float* p = in + (size_t)token * CC;
    float v[6];
    float s = 0.f;
#pragma unroll
    for (int j = 0; j < 6; ++j) { v[j] = p[lane + 32 * j]; s += v[j]; }
    const float mean = red_sum32(s) * (1.0f / CC);
    float s2 = 0.f;
#pragma unroll
    for (int j = 0; j < 6; ++j) { float d = v[j] - mean; s2 += d * d; }
    const float rs = rsqrtf(red_sum32(s2) * (1.0f / CC) + 1e-5f);

    int drow = token;
    if constexpr (REMAP) {
        // cyclic shift (-3,-3) then 7x7 window partition
        const int bb = token / (CH * CW);
        const int l  = token - bb * (CH * CW);
        int h = l / CW, w = l - (l / CW) * CW;
        int hp = h - 3; if (hp < 0) hp += CH;
        int wp = w - 3; if (wp < 0) wp += CW;
        const int wi = hp / 7, ti = hp - (hp / 7) * 7;
        const int wj = wp / 7, tj = wp - (wp / 7) * 7;
        drow = ((bb * CNWIN + wi * 8 + wj) * CN) + ti * 7 + tj;
    }
    _Float16* q = out + (size_t)drow * CC;
#pragma unroll
    for (int j = 0; j < 6; ++j) {
        const int c = lane + 32 * j;
        q[c] = (_Float16)((v[j] - mean) * rs * g[c] + b[c]);
    }
}

// ---------------------------------------------------------------- generic WMMA GEMM: out = A[M,K]f16 @ Wt[N,K]^T + bias
// EPI 0: f16 store                 (qkv)
// EPI 1: f16 store + exact GELU    (fc1)
// EPI 2: f32 store + residual      (fc2 -> final out)
// EPI 3: f32 store + residual with window-reverse+unshift row remap (proj -> x_res)
template<int EPI, int NN, int KK>
__global__ __launch_bounds__(256) void swin_gemm_wmma(const _Float16* __restrict__ A,
                                                      const _Float16* __restrict__ Wt,
                                                      const float* __restrict__ bias,
                                                      _Float16* __restrict__ outh,
                                                      float* __restrict__ outf,
                                                      const float* __restrict__ resid) {
    const int lane = threadIdx.x & 31;
    const int wv   = threadIdx.x >> 5;
    const int row0 = blockIdx.y * 128 + wv * 16;
    const int col0 = blockIdx.x * 64;

    v8f acc[4] = {};
#pragma unroll
    for (int k0 = 0; k0 < KK; k0 += 32) {
        v16h a = load_a_frag(A, KK, row0, k0, lane);
#pragma unroll
        for (int nt = 0; nt < 4; ++nt) {
            v16h bfr = load_b_frag(Wt, KK, col0 + nt * 16, k0, lane);
            acc[nt] = wmma16x16x32(a, bfr, acc[nt]);
        }
    }

    const int hh = lane >> 4, l15 = lane & 15;
    int drow[8];
#pragma unroll
    for (int r = 0; r < 8; ++r) drow[r] = row0 + r + 8 * hh;
    if constexpr (EPI == 3) {
#pragma unroll
        for (int r = 0; r < 8; ++r) {
            const int rw  = drow[r];
            const int win = rw / CN, t = rw - win * CN;
            const int bb  = win >> 6, wim = win & 63;
            const int wi  = wim >> 3, wj = wim & 7;
            const int ti  = t / 7,  tj = t - (t / 7) * 7;
            int h = wi * 7 + ti + 3; if (h >= CH) h -= CH;   // reverse roll (+3,+3)
            int w = wj * 7 + tj + 3; if (w >= CW) w -= CW;
            drow[r] = bb * (CH * CW) + h * CW + w;
        }
    }

#pragma unroll
    for (int nt = 0; nt < 4; ++nt) {
        const int col = col0 + nt * 16 + l15;
        const float bv = bias[col];
#pragma unroll
        for (int r = 0; r < 8; ++r) {
            float val = acc[nt][r] + bv;
            if constexpr (EPI == 0) {
                outh[(size_t)(row0 + r + 8 * hh) * NN + col] = (_Float16)val;
            } else if constexpr (EPI == 1) {
                val = 0.5f * val * (1.0f + erff(val * 0.70710678118654752f));
                outh[(size_t)(row0 + r + 8 * hh) * NN + col] = (_Float16)val;
            } else if constexpr (EPI == 2) {
                const size_t i = (size_t)(row0 + r + 8 * hh) * NN + col;
                outf[i] = val + resid[i];
            } else {
                const size_t i = (size_t)drow[r] * NN + col;
                outf[i] = val + resid[i];
            }
        }
    }
}

// ---------------------------------------------------------------- shifted-window attention: one block (4 waves) per (window, head)
__global__ __launch_bounds__(128) void swin_attn_wmma(const _Float16* __restrict__ qkv,
                                                      const float* __restrict__ fused,
                                                      _Float16* __restrict__ obuf) {
    __shared__ alignas(32) _Float16 qs[64 * 32];
    __shared__ alignas(32) _Float16 ks[64 * 32];
    __shared__ alignas(32) _Float16 vt[32 * 64];   // V transposed: [dim][key]
    __shared__ alignas(32) _Float16 ps[64 * 64];   // softmax probs

    const int win  = blockIdx.x / CNH;
    const int head = blockIdx.x - win * CNH;
    const int tid  = threadIdx.x;
    const int lane = tid & 31, wv = tid >> 5;

    const _Float16* base = qkv + (size_t)win * CN * CQKV + head * CHD;
    for (int i = tid; i < 64 * 32; i += 128) {
        const int t = i >> 5, d = i & 31;
        float qv = 0.f, kv = 0.f, vv = 0.f;
        if (t < CN) {
            const _Float16* rp = base + (size_t)t * CQKV;
            qv = (float)rp[d] * 0.17677669529663689f;   // 1/sqrt(32)
            kv = (float)rp[CC + d];
            vv = (float)rp[2 * CC + d];
        }
        qs[t * 32 + d] = (_Float16)qv;
        ks[t * 32 + d] = (_Float16)kv;
        vt[d * 64 + t] = (_Float16)vv;
    }
    __syncthreads();

    // S = q @ k^T  (K = 32, single WMMA step; wave wv owns rows [16*wv, 16*wv+16))
    v8f acc[4] = {};
    {
        v16h a = load_a_frag(qs, 32, wv * 16, 0, lane);
#pragma unroll
        for (int nt = 0; nt < 4; ++nt) {
            v16h bfr = load_b_frag(ks, 32, nt * 16, 0, lane);
            acc[nt] = wmma16x16x32(a, bfr, acc[nt]);
        }
    }

    // + fused(bias+mask, padded with -1e30): branch-free coalesced adds
    const int hh = lane >> 4, l15 = lane & 15;
    const float* fb = fused + (((size_t)(win & 63) * CNH + head) << 12);  // * 64*64
#pragma unroll
    for (int nt = 0; nt < 4; ++nt) {
#pragma unroll
        for (int r = 0; r < 8; ++r) {
            const int m = wv * 16 + r + 8 * hh;
            const int n = nt * 16 + l15;
            acc[nt][r] += fb[m * 64 + n];
        }
    }

    // row softmax: row m lives in one 16-lane half (l15 sweeps n), 4 values/lane
#pragma unroll
    for (int r = 0; r < 8; ++r) {
        float mx = fmaxf(fmaxf(acc[0][r], acc[1][r]), fmaxf(acc[2][r], acc[3][r]));
        mx = red_max16(mx);
        float sum = 0.f;
#pragma unroll
        for (int nt = 0; nt < 4; ++nt) { float e = __expf(acc[nt][r] - mx); acc[nt][r] = e; sum += e; }
        sum = red_sum16(sum);
        const float inv = 1.0f / sum;
#pragma unroll
        for (int nt = 0; nt < 4; ++nt) acc[nt][r] *= inv;
    }

    // round-trip P through LDS to re-stripe into A-fragment layout
#pragma unroll
    for (int nt = 0; nt < 4; ++nt)
#pragma unroll
        for (int r = 0; r < 8; ++r)
            ps[(wv * 16 + r + 8 * hh) * 64 + nt * 16 + l15] = (_Float16)acc[nt][r];
    __syncthreads();

    // O = P @ V   (M=16/wave, N=32, K=64 -> two K-steps, two N-tiles)
    v8f oacc[2] = {};
#pragma unroll
    for (int kstep = 0; kstep < 2; ++kstep) {
        v16h pa = load_a_frag(ps, 64, wv * 16, kstep * 32, lane);
#pragma unroll
        for (int nt = 0; nt < 2; ++nt) {
            v16h vb = load_b_frag(vt, 64, nt * 16, kstep * 32, lane);
            oacc[nt] = wmma16x16x32(pa, vb, oacc[nt]);
        }
    }
#pragma unroll
    for (int nt = 0; nt < 2; ++nt)
#pragma unroll
        for (int r = 0; r < 8; ++r) {
            const int m = wv * 16 + r + 8 * hh;
            if (m < CN)
                obuf[((size_t)win * CN + m) * CC + head * CHD + nt * 16 + l15] = (_Float16)oacc[nt][r];
        }
}

// ---------------------------------------------------------------- launch
extern "C" void kernel_launch(void* const* d_in, const int* in_sizes, int n_in,
                              void* d_out, int out_size, void* d_ws, size_t ws_size,
                              hipStream_t stream) {
    (void)in_sizes; (void)n_in; (void)out_size; (void)ws_size;
    const float* x      = (const float*)d_in[0];
    const float* n1g    = (const float*)d_in[1];
    const float* n1b    = (const float*)d_in[2];
    const float* qkv_w  = (const float*)d_in[3];
    const float* qkv_b  = (const float*)d_in[4];
    const float* proj_w = (const float*)d_in[5];
    const float* proj_b = (const float*)d_in[6];
    const float* n2g    = (const float*)d_in[7];
    const float* n2b    = (const float*)d_in[8];
    const float* fc1_w  = (const float*)d_in[9];
    const float* fc1_b  = (const float*)d_in[10];
    const float* fc2_w  = (const float*)d_in[11];
    const float* fc2_b  = (const float*)d_in[12];
    const float* rel_t  = (const float*)d_in[13];
    const int*   rel_i  = (const int*)d_in[14];
    const float* maskm  = (const float*)d_in[15];
    float* outp = (float*)d_out;

    char* ws = (char*)d_ws;
    // sizes (bytes), all 256-aligned
    const size_t SZ_XW  = (size_t)CROWS * CC * 2;     //  38,535,168  (ln1 out, reused for ln2 out)
    const size_t SZ_QKV = (size_t)CROWS * CQKV * 2;   // 115,605,504
    const size_t SZ_O   = (size_t)CROWS * CC * 2;     //  38,535,168
    _Float16* xw     = (_Float16*)(ws);
    _Float16* qkvbuf = (_Float16*)(ws + SZ_XW);
    _Float16* obuf   = (_Float16*)(ws + SZ_XW + SZ_QKV);
    _Float16* h1     = (_Float16*)(ws + SZ_XW);                 // aliases qkv+o (both dead by fc1)
    float*    xres   = (float*)   (ws + SZ_XW + SZ_QKV + SZ_O);
    char*     wbase  = ws + SZ_XW + SZ_QKV + SZ_O + (size_t)CROWS * CC * 4;
    _Float16* qkv_wt = (_Float16*)(wbase);
    _Float16* proj_wt= (_Float16*)(wbase + 221184);
    _Float16* fc1_wt = (_Float16*)(wbase + 221184 + 73728);
    _Float16* fc2_wt = (_Float16*)(wbase + 221184 + 73728 + 294912);
    float*    fused  = (float*)   (wbase + 221184 + 73728 + 294912 + 294912);  // 64*6*64*64 f32

    // f16 transposed weights + fused bias/mask table
    swin_transpose_f16<<<(CC * CQKV + 255) / 256, 256, 0, stream>>>(qkv_w, qkv_wt, CC, CQKV);
    swin_transpose_f16<<<(CC * CC + 255) / 256, 256, 0, stream>>>(proj_w, proj_wt, CC, CC);
    swin_transpose_f16<<<(CC * CHID + 255) / 256, 256, 0, stream>>>(fc1_w, fc1_wt, CC, CHID);
    swin_transpose_f16<<<(CHID * CC + 255) / 256, 256, 0, stream>>>(fc2_w, fc2_wt, CHID, CC);
    swin_fuse_bias_mask<<<(CNWIN * CNH * 64 * 64) / 256, 256, 0, stream>>>(rel_t, rel_i, maskm, fused);

    // LN1 + shift + window partition -> xw [100352, 192] f16
    swin_ln_f16<true><<<CROWS / 8, 256, 0, stream>>>(x, n1g, n1b, xw);
    // QKV: [100352,192] @ [192,576] -> qkv f16
    swin_gemm_wmma<0, CQKV, CC><<<dim3(CQKV / 64, CROWS / 128), 256, 0, stream>>>(
        xw, qkv_wt, qkv_b, qkvbuf, nullptr, nullptr);
    // attention per (window, head)
    swin_attn_wmma<<<(CROWS / CN) * CNH, 128, 0, stream>>>(qkvbuf, fused, obuf);
    // proj + window-reverse/unshift + residual(x) -> xres f32
    swin_gemm_wmma<3, CC, CC><<<dim3(CC / 64, CROWS / 128), 256, 0, stream>>>(
        obuf, proj_wt, proj_b, nullptr, xres, x);
    // LN2 -> y (reuse xw)
    swin_ln_f16<false><<<CROWS / 8, 256, 0, stream>>>(xres, n2g, n2b, xw);
    // fc1 + GELU -> h1 f16
    swin_gemm_wmma<1, CHID, CC><<<dim3(CHID / 64, CROWS / 128), 256, 0, stream>>>(
        xw, fc1_wt, fc1_b, h1, nullptr, nullptr);
    // fc2 + residual(xres) -> final f32 output
    swin_gemm_wmma<2, CC, CHID><<<dim3(CC / 64, CROWS / 128), 256, 0, stream>>>(
        h1, fc2_wt, fc2_b, nullptr, outp, xres);
}
